// DiTBlock_23210003267769
// MI455X (gfx1250) — compile-verified, matrix operand order checked by reading, not run on hardware
//
#include <hip/hip_runtime.h>
#include <hip/hip_bf16.h>
#include <math.h>

// ---------------------------------------------------------------------------
// Types for CDNA5 WMMA / TDM (gfx1250, wave32)
// ---------------------------------------------------------------------------
typedef __attribute__((ext_vector_type(16))) __bf16 v16bf;
typedef __attribute__((ext_vector_type(8)))  __bf16 v8bf;
typedef __attribute__((ext_vector_type(8)))  float  v8f;
typedef __attribute__((ext_vector_type(4)))  unsigned int v4u;
typedef __attribute__((ext_vector_type(8)))  int v8i;
typedef __attribute__((ext_vector_type(4)))  int v4i;

#if defined(__has_builtin)
#if __has_builtin(__builtin_amdgcn_tensor_load_to_lds) && \
    __has_builtin(__builtin_amdgcn_s_wait_tensorcnt)
#define USE_TDM 1
#endif
#endif
#ifndef USE_TDM
#define USE_TDM 0
#endif

static __device__ __forceinline__ v16bf cat8(v8bf lo, v8bf hi) {
    return __builtin_shufflevector(lo, hi, 0,1,2,3,4,5,6,7,8,9,10,11,12,13,14,15);
}
static __device__ __forceinline__ v8f wmma_bf16(v16bf a, v16bf b, v8f c) {
    // D = A(16x32 bf16) * B(32x16 bf16) + C(16x16 f32)
    return __builtin_amdgcn_wmma_f32_16x16x32_bf16(false, a, false, b, (short)0, c, false, false);
}

// Problem constants
constexpr int BB    = 4;
constexpr int NN    = 1024;
constexpr int HH    = 1024;
constexpr int HEADS = 16;
constexpr int DH    = 64;
constexpr int DFF   = 4096;
constexpr int MROWS = BB * NN;   // 4096
constexpr int SIXH  = 6 * HH;    // 6144

#if USE_TDM
// ---------------------------------------------------------------------------
// Issue one TDM load: 128x32-element bf16 tile of a row-major [dim1 x dim0]
// tensor into LDS, with HW padding of 4 DWORDs every 16 DWORDs -> LDS row
// stride of 40 halfs (matches the WMMA fragment loader below).
// D# group0: count=1 | lds_addr | 57b global_addr | type=2
// D# group1: data_size=2B, pad_enable, pad_interval=3 (16 DW), pad_amount=3
//            (4 DW), tensor_dim0/1, tile 32x128, 48b dim0 stride.
// This toolchain exposes the 6-arg builtin (g0, g1, g2, g3, g4, cpol).
// ---------------------------------------------------------------------------
static __device__ __forceinline__ void tdm_load_tile(
        unsigned ldsAddr, unsigned long long gAddr,
        unsigned dim0, unsigned dim1, unsigned stride0) {
    v4u g0;
    g0.x = 1u;                                     // count = 1 descriptor
    g0.y = ldsAddr;                                // LDS byte address
    g0.z = (unsigned)(gAddr & 0xFFFFFFFFull);      // global addr [31:0]
    g0.w = (unsigned)((gAddr >> 32) & 0x1FFFFFFull) | (2u << 30);  // [56:32]|type=2
    v8i g1;
    g1[0] = (int)((1u << 16) | (1u << 20) | (3u << 22) | (3u << 25));
    g1[1] = (int)((dim0 & 0xFFFFu) << 16);                     // dim0[15:0]
    g1[2] = (int)((dim0 >> 16) | ((dim1 & 0xFFFFu) << 16));    // dim0[31:16]|dim1[15:0]
    g1[3] = (int)((dim1 >> 16) | (32u << 16));                 // dim1[31:16]|tile0=32
    g1[4] = (int)128;                                          // tile1=128, tile2=0
    g1[5] = (int)stride0;                                      // stride0[31:0]
    g1[6] = 0;                                                 // stride0[47:32]|stride1 lo
    g1[7] = 0;
    v4i z4 = {};
    v8i z8 = {};
    __builtin_amdgcn_tensor_load_to_lds(g0, g1, z4, z4, z8, 0);
}
#endif

// ---------------------------------------------------------------------------
// fp32 -> bf16 transpose:  in[K x N] (row-major) -> out[N x K]
// ---------------------------------------------------------------------------
__global__ void k_f2bf_t(const float* __restrict__ in, __bf16* __restrict__ out,
                         int K, int N) {
    int id = blockIdx.x * blockDim.x + threadIdx.x;
    if (id >= K * N) return;
    int k = id / N, n = id % N;                  // consecutive id -> coalesced read
    out[(size_t)n * K + k] = (__bf16)in[id];
}

// ---------------------------------------------------------------------------
// cm = silu(c) @ w_mod + b_mod    [B, 6H]
// ---------------------------------------------------------------------------
__global__ void k_cond(const float* __restrict__ c, const float* __restrict__ wmod,
                       const float* __restrict__ bmod, float* __restrict__ cm) {
    int id = blockIdx.x * blockDim.x + threadIdx.x;
    if (id >= BB * SIXH) return;
    int b = id / SIXH, j = id % SIXH;
    const float* cb = c + b * HH;
    float s = bmod[j];
    for (int k = 0; k < HH; ++k) {
        float cv = cb[k];
        s += (cv / (1.0f + __expf(-cv))) * wmod[(size_t)k * SIXH + j];
    }
    cm[id] = s;
}

// ---------------------------------------------------------------------------
// LayerNorm + adaLN modulate -> bf16. One 256-thread block per token row.
// ---------------------------------------------------------------------------
__global__ void __launch_bounds__(256)
k_ln_mod(const float* __restrict__ x, const float* __restrict__ cm,
         int shiftOff, int scaleOff, __bf16* __restrict__ out) {
    __shared__ float red[256];
    const int row = blockIdx.x;
    const int b   = row >> 10;
    const int tid = threadIdx.x;
    const float* xr = x + (size_t)row * HH;

    float lv[4]; float s = 0.0f;
#pragma unroll
    for (int i = 0; i < 4; ++i) { lv[i] = xr[tid + i * 256]; s += lv[i]; }
    red[tid] = s; __syncthreads();
    for (int off = 128; off > 0; off >>= 1) { if (tid < off) red[tid] += red[tid + off]; __syncthreads(); }
    const float mu = red[0] * (1.0f / (float)HH);
    __syncthreads();

    float s2 = 0.0f;
#pragma unroll
    for (int i = 0; i < 4; ++i) { float d = lv[i] - mu; s2 += d * d; }
    red[tid] = s2; __syncthreads();
    for (int off = 128; off > 0; off >>= 1) { if (tid < off) red[tid] += red[tid + off]; __syncthreads(); }
    const float rinv = rsqrtf(red[0] * (1.0f / (float)HH) + 1e-5f);

    const float* cmb = cm + (size_t)b * SIXH;
#pragma unroll
    for (int i = 0; i < 4; ++i) {
        int col = tid + i * 256;
        float sh = cmb[shiftOff + col], sc = cmb[scaleOff + col];
        out[(size_t)row * HH + col] = (__bf16)(((lv[i] - mu) * rinv) * (1.0f + sc) + sh);
    }
}

// ---------------------------------------------------------------------------
// Tiled bf16 WMMA GEMM:  C[M,N] = A[M,K] @ Bt[N,K]^T (+bias, epilogue)
// A row-major [M][K]; Bt is the transposed weight [N][K], so BOTH tiles are
// row-major 128x32 and are DMA'd by the Tensor Data Mover with HW padding
// (LDS stride 40 halfs), double-buffered with s_wait_tensorcnt.
// 8 waves: 4 M-slabs x 2 N-slabs, each wave 2x4 WMMA tiles per k-step.
// Epilogues: 0=QKV(bias->bf16) 1=PROJ(bias,(1+gate)->f32)
//            2=MLP1(bias,geluExact->bf16) 3=MLP2(bias,resid+gate*v->f32)
// ---------------------------------------------------------------------------
template <int EPI>
__global__ void __launch_bounds__(256)
k_gemm(const __bf16* __restrict__ A, const __bf16* __restrict__ Bt,
       const float* __restrict__ bias, const float* __restrict__ cmod,
       const float* __restrict__ resid,
       float* __restrict__ outF, __bf16* __restrict__ outBF,
       int M, int N, int K) {
    constexpr int LDA = 40;                 // 32 + 8 halfs pad
    constexpr int TILE = 128 * LDA;         // halfs per tile
    __shared__ __bf16 smem[4 * TILE];       // A0 B0 A1 B1 (double buffered)

    const int tid  = threadIdx.x;
    const int lane = tid & 31;
    const int wid  = tid >> 5;
    const int wm   = wid & 3;
    const int wn   = wid >> 2;
    const int lr   = lane & 15;
    const int half = lane >> 4;
    const int kb   = half * 8;
    const int bm   = blockIdx.y * 128;
    const int bn   = blockIdx.x * 128;
    const int T    = K >> 5;                // k-tiles of 32

    v8f acc[2][4] = {};

#if USE_TDM
    const unsigned long long Aga0 = (unsigned long long)(uintptr_t)A  + (size_t)bm * K * 2;
    const unsigned long long Bga0 = (unsigned long long)(uintptr_t)Bt + (size_t)bn * K * 2;
    const unsigned smemBase = (unsigned)(uintptr_t)smem;
    if (wid == 0) {   // prime the pipeline: tile 0 into buffer 0
        tdm_load_tile(smemBase,            Aga0, K, M, K);
        tdm_load_tile(smemBase + TILE * 2, Bga0, K, N, K);
    }
#endif

    for (int t = 0; t < T; ++t) {
        const int buf = t & 1;
        __bf16* As = smem + buf * 2 * TILE;
        __bf16* Bs = As + TILE;

#if USE_TDM
        if (wid == 0) {
            if (t + 1 < T) {    // issue next tile into the other buffer
                const int nb = (t + 1) & 1;
                unsigned ldsA = smemBase + nb * 2 * TILE * 2;
                tdm_load_tile(ldsA,            Aga0 + (size_t)(t + 1) * 64, K, M, K);
                tdm_load_tile(ldsA + TILE * 2, Bga0 + (size_t)(t + 1) * 64, K, N, K);
                __builtin_amdgcn_s_wait_tensorcnt(2);  // tile t complete
            } else {
                __builtin_amdgcn_s_wait_tensorcnt(0);
            }
        }
        __syncthreads();
#else
        // manual staging fallback (no TDM builtin on this toolchain)
        const int k0 = t * 32;
#pragma unroll
        for (int i = 0; i < 2; ++i) {
            int cc = tid + i * 256;
            int r = cc >> 2, kc = (cc & 3) << 3;
            *(v8bf*)(&As[r * LDA + kc]) =
                *(const v8bf*)(A + (size_t)(bm + r) * K + k0 + kc);
            *(v8bf*)(&Bs[r * LDA + kc]) =
                *(const v8bf*)(Bt + (size_t)(bn + r) * K + k0 + kc);
        }
        __syncthreads();
#endif

        v16bf af[2], bfr[4];
#pragma unroll
        for (int i = 0; i < 2; ++i) {
            const __bf16* p = &As[(wm * 32 + i * 16 + lr) * LDA + kb];
            af[i] = cat8(*(const v8bf*)p, *(const v8bf*)(p + 16));
        }
#pragma unroll
        for (int j = 0; j < 4; ++j) {
            const __bf16* p = &Bs[(wn * 64 + j * 16 + lr) * LDA + kb];
            bfr[j] = cat8(*(const v8bf*)p, *(const v8bf*)(p + 16));
        }
#pragma unroll
        for (int i = 0; i < 2; ++i)
#pragma unroll
            for (int j = 0; j < 4; ++j)
                acc[i][j] = wmma_bf16(af[i], bfr[j], acc[i][j]);
        __syncthreads();
    }

    // ---- fused epilogue ----
#pragma unroll
    for (int i = 0; i < 2; ++i) {
#pragma unroll
        for (int j = 0; j < 4; ++j) {
            const int r0  = bm + wm * 32 + i * 16 + half * 8;
            const int col = bn + wn * 64 + j * 16 + lr;
            const float bv = bias[col];
#pragma unroll
            for (int r = 0; r < 8; ++r) {
                const int row = r0 + r;
                float v = acc[i][j][r] + bv;
                if (EPI == 0) {
                    outBF[(size_t)row * N + col] = (__bf16)v;
                } else if (EPI == 1) {               // x2 = v*(1+gate_msa)
                    int b = row >> 10;
                    float g = cmod[(size_t)b * SIXH + 2 * HH + col];
                    outF[(size_t)row * HH + col] = v * (1.0f + g);
                } else if (EPI == 2) {               // exact GELU
                    float gl = 0.5f * v * (1.0f + erff(v * 0.70710678118654752f));
                    outBF[(size_t)row * N + col] = (__bf16)gl;
                } else {                             // out = x2 + gate_mlp*v
                    int b = row >> 10;
                    float g = cmod[(size_t)b * SIXH + 5 * HH + col];
                    outF[(size_t)row * HH + col] =
                        resid[(size_t)row * HH + col] + g * v;
                }
            }
        }
    }
}

// ---------------------------------------------------------------------------
// Transpose V out of the QKV buffer: vt[b,head,d,key] (bf16) so attention's
// P@V B-fragments are two contiguous 16B global loads per lane.
// ---------------------------------------------------------------------------
__global__ void k_vt(const __bf16* __restrict__ qkv, __bf16* __restrict__ vt) {
    int id = blockIdx.x * blockDim.x + threadIdx.x;   // coalesced writes
    if (id >= BB * HEADS * DH * NN) return;
    int key  = id & (NN - 1);
    int rest = id >> 10;
    int d    = rest & (DH - 1);
    int bh   = rest >> 6;
    int b    = bh >> 4, head = bh & 15;
    vt[id] = qkv[(size_t)(b * NN + key) * (3 * HH) + 2 * HH + head * DH + d];
}

// ---------------------------------------------------------------------------
// Flash attention. Grid (B*HEADS, N/128), 256 threads = 8 waves; each wave
// owns 16 query rows, streams keys in chunks of 32. Q/K fragments from
// global; V fragments from the pre-transposed vt (contiguous); P converts
// C-layout -> A-layout via a wave-private LDS tile. No block barriers in the
// main loop. Online softmax via 16-lane shfl_xor (halves == C-layout rows).
// ---------------------------------------------------------------------------
__global__ void __launch_bounds__(256)
k_attn(const __bf16* __restrict__ qkv, const __bf16* __restrict__ vt,
       __bf16* __restrict__ outp) {
    constexpr int LDT = 40;
    __shared__ __bf16 Pj[8][16 * LDT];

    const int b    = blockIdx.x >> 4;
    const int head = blockIdx.x & 15;
    const int tid  = threadIdx.x;
    const int lane = tid & 31;
    const int wid  = tid >> 5;
    const int lr   = lane & 15;
    const int half = lane >> 4;
    const int kb   = half * 8;
    const int qr   = blockIdx.y * 128 + wid * 16;
    const size_t RS = 3 * HH;

    const __bf16* qbase = qkv + ((size_t)(b * NN + qr + lr)) * RS + head * DH;
    v16bf qf[2];
#pragma unroll
    for (int f = 0; f < 2; ++f) {
        const __bf16* p = qbase + f * 32 + kb;
        qf[f] = cat8(*(const v8bf*)p, *(const v8bf*)(p + 16));
    }

    v8f ov[4] = {};
    float mrun[8], lsum[8];
#pragma unroll
    for (int r = 0; r < 8; ++r) { mrun[r] = -1e30f; lsum[r] = 0.0f; }

    const __bf16* kbaseP = qkv + ((size_t)(b * NN)) * RS + HH + head * DH;
    const __bf16* vbaseP = vt + ((size_t)(b * HEADS + head)) * DH * NN;

    for (int kc = 0; kc < NN; kc += 32) {
        v8f s[2] = {};
#pragma unroll
        for (int j = 0; j < 2; ++j) {
            const __bf16* kp = kbaseP + (size_t)(kc + j * 16 + lr) * RS;
#pragma unroll
            for (int f = 0; f < 2; ++f) {
                const __bf16* p = kp + f * 32 + kb;
                v16bf kf = cat8(*(const v8bf*)p, *(const v8bf*)(p + 16));
                s[j] = wmma_bf16(qf[f], kf, s[j]);
            }
        }
#pragma unroll
        for (int j = 0; j < 2; ++j)
#pragma unroll
            for (int r = 0; r < 8; ++r) s[j][r] *= 0.125f;

        float mnew[8];
#pragma unroll
        for (int r = 0; r < 8; ++r) {
            float v = fmaxf(s[0][r], s[1][r]);
#pragma unroll
            for (int off = 1; off < 16; off <<= 1) v = fmaxf(v, __shfl_xor(v, off, 32));
            mnew[r] = fmaxf(mrun[r], v);
        }
#pragma unroll
        for (int r = 0; r < 8; ++r) {
            float f = __expf(mrun[r] - mnew[r]);
            lsum[r] *= f; mrun[r] = mnew[r];
#pragma unroll
            for (int t = 0; t < 4; ++t) ov[t][r] *= f;
        }
#pragma unroll
        for (int j = 0; j < 2; ++j)
#pragma unroll
            for (int r = 0; r < 8; ++r) s[j][r] = __expf(s[j][r] - mrun[r]);
#pragma unroll
        for (int r = 0; r < 8; ++r) {
            float rs = s[0][r] + s[1][r];
#pragma unroll
            for (int off = 1; off < 16; off <<= 1) rs += __shfl_xor(rs, off, 32);
            lsum[r] += rs;
        }

        __bf16* pw = Pj[wid];
#pragma unroll
        for (int j = 0; j < 2; ++j)
#pragma unroll
            for (int r = 0; r < 8; ++r)
                pw[(r + half * 8) * LDT + j * 16 + lr] = (__bf16)s[j][r];
        const __bf16* pp = pw + lr * LDT + kb;
        v16bf pf = cat8(*(const v8bf*)pp, *(const v8bf*)(pp + 16));

#pragma unroll
        for (int t = 0; t < 4; ++t) {
            const __bf16* vp = vbaseP + (size_t)(t * 16 + lr) * NN + kc + kb;
            v16bf vf = cat8(*(const v8bf*)vp, *(const v8bf*)(vp + 16));
            ov[t] = wmma_bf16(pf, vf, ov[t]);
        }
    }

#pragma unroll
    for (int t = 0; t < 4; ++t)
#pragma unroll
        for (int r = 0; r < 8; ++r) {
            float val = ov[t][r] / lsum[r];
            size_t row = (size_t)(b * NN + qr + r + half * 8);
            outp[row * HH + head * DH + t * 16 + lr] = (__bf16)val;
        }
}

// ---------------------------------------------------------------------------
// Host-side orchestration
// ---------------------------------------------------------------------------
extern "C" void kernel_launch(void* const* d_in, const int* in_sizes, int n_in,
                              void* d_out, int out_size, void* d_ws, size_t ws_size,
                              hipStream_t stream) {
    (void)in_sizes; (void)n_in; (void)out_size; (void)ws_size;

    const float* x      = (const float*)d_in[0];
    const float* c      = (const float*)d_in[1];
    const float* w_mod  = (const float*)d_in[2];
    const float* b_mod  = (const float*)d_in[3];
    const float* w_qkv  = (const float*)d_in[4];
    const float* b_qkv  = (const float*)d_in[5];
    const float* w_proj = (const float*)d_in[6];
    const float* b_proj = (const float*)d_in[7];
    const float* w1     = (const float*)d_in[8];
    const float* b1     = (const float*)d_in[9];
    const float* w2     = (const float*)d_in[10];
    const float* b2     = (const float*)d_in[11];
    float* out = (float*)d_out;

    char* ws = (char*)d_ws;
    size_t off = 0;
    auto alloc = [&](size_t bytes) -> char* {
        char* p = ws + off;
        off += (bytes + 255) & ~(size_t)255;
        return p;
    };
    __bf16* wqkvT  = (__bf16*)alloc((size_t)HH * 3 * HH * 2);   // [3H][H]
    __bf16* wprojT = (__bf16*)alloc((size_t)HH * HH * 2);       // [H][H]
    __bf16* w1T    = (__bf16*)alloc((size_t)HH * DFF * 2);      // [DFF][H]
    __bf16* w2T    = (__bf16*)alloc((size_t)DFF * HH * 2);      // [H][DFF]
    float*  cm     = (float*)alloc((size_t)BB * SIXH * 4);
    __bf16* xm_bf  = (__bf16*)alloc((size_t)MROWS * HH * 2);
    __bf16* qkv_bf = (__bf16*)alloc((size_t)MROWS * 3 * HH * 2);
    __bf16* vtg    = (__bf16*)alloc((size_t)MROWS * HH * 2);    // [b,h,d,key]
    __bf16* attn_bf= (__bf16*)alloc((size_t)MROWS * HH * 2);
    float*  x2     = (float*)alloc((size_t)MROWS * HH * 4);
    __bf16* xm2_bf = (__bf16*)alloc((size_t)MROWS * HH * 2);
    __bf16* hid_bf = (__bf16*)alloc((size_t)MROWS * DFF * 2);

    dim3 blk(256);

    // 1) transpose-convert weights to bf16 [N][K]
    {
        int n;
        n = HH * 3 * HH; k_f2bf_t<<<(n + 255) / 256, blk, 0, stream>>>(w_qkv,  wqkvT,  HH, 3 * HH);
        n = HH * HH;     k_f2bf_t<<<(n + 255) / 256, blk, 0, stream>>>(w_proj, wprojT, HH, HH);
        n = HH * DFF;    k_f2bf_t<<<(n + 255) / 256, blk, 0, stream>>>(w1,     w1T,    HH, DFF);
        n = DFF * HH;    k_f2bf_t<<<(n + 255) / 256, blk, 0, stream>>>(w2,     w2T,    DFF, HH);
    }
    // 2) modulation vector
    k_cond<<<(BB * SIXH + 255) / 256, blk, 0, stream>>>(c, w_mod, b_mod, cm);
    // 3) LN + modulate (MSA)
    k_ln_mod<<<MROWS, blk, 0, stream>>>(x, cm, 0, HH, xm_bf);
    // 4) QKV GEMM
    k_gemm<0><<<dim3(3 * HH / 128, MROWS / 128), blk, 0, stream>>>(
        xm_bf, wqkvT, b_qkv, nullptr, nullptr, nullptr, qkv_bf, MROWS, 3 * HH, HH);
    // 5) V transpose + flash attention
    k_vt<<<(MROWS * HH + 255) / 256, blk, 0, stream>>>(qkv_bf, vtg);
    k_attn<<<dim3(BB * HEADS, NN / 128), blk, 0, stream>>>(qkv_bf, vtg, attn_bf);
    // 6) proj GEMM + (1+gate_msa) -> x2
    k_gemm<1><<<dim3(HH / 128, MROWS / 128), blk, 0, stream>>>(
        attn_bf, wprojT, b_proj, cm, nullptr, x2, nullptr, MROWS, HH, HH);
    // 7) LN + modulate (MLP)
    k_ln_mod<<<MROWS, blk, 0, stream>>>(x2, cm, 3 * HH, 4 * HH, xm2_bf);
    // 8) MLP1 GEMM + GELU
    k_gemm<2><<<dim3(DFF / 128, MROWS / 128), blk, 0, stream>>>(
        xm2_bf, w1T, b1, nullptr, nullptr, nullptr, hid_bf, MROWS, DFF, HH);
    // 9) MLP2 GEMM + residual + gate_mlp -> out
    k_gemm<3><<<dim3(HH / 128, MROWS / 128), blk, 0, stream>>>(
        hid_bf, w2T, b2, cm, x2, out, nullptr, MROWS, HH, DFF);
}
